// MambaMLPBlock_36627481100485
// MI455X (gfx1250) — matrix operand order repaired and checked by reading, not compiled, not run on hardware
//
#include <hip/hip_runtime.h>
#include <hip/hip_bf16.h>
#include <math.h>

// ---------------------------------------------------------------------------
// MI455X (gfx1250) fused Mamba+MLP block, round 2.
// All GEMMs: v_wmma_f32_16x16x32_bf16, fp32 accum.
// Weights + LDS A-tiles stored in WMMA fragment-swizzled order so every
// fragment load is a contiguous 32B vector load (global_load_b128 / ds_load_b128)
// instead of 16 scattered u16 loads.
// ---------------------------------------------------------------------------

typedef __attribute__((ext_vector_type(16))) __bf16 v16bf;
typedef __attribute__((ext_vector_type(8)))  __bf16 v8bf;
typedef __attribute__((ext_vector_type(8)))  float  v8f;
typedef __bf16 bf16_t;

#define BSZ   4
#define TLEN  256
#define NSP   64
#define CDIM  128
#define DIN   256
#define DST   16
#define DTR   8
#define XPAD  48              // 40 (dt_rank + 2*d_state) padded to 3x16 tiles
#define SEQ   (BSZ * NSP)     // 256 sequences
#define TOK   (SEQ * TLEN)    // 65536 tokens

// ---- fragment swizzle maps (CDNA5 ISA 7.12.2, 16-bit A 16x32 / B 32x16) ----
// A element (row m, kin) lives at lane = m + ((kin>>3)&1)*16,
//                                 elem = (kin&7) + ((kin>>4)<<3)
__device__ __forceinline__ int a_lane(int row, int kin) { return row + (((kin >> 3) & 1) << 4); }
__device__ __forceinline__ int a_elem(int kin)          { return (kin & 7) + ((kin >> 4) << 3); }
// B element (kin, n) lives at lane = n + ((kin>>3)&1)*16, elem = (kin&7) + ((kin>>4)&1)*8
__device__ __forceinline__ size_t b_swz(int k, int n, int NT) {
  int kt = k >> 5, nt = n >> 4;
  int lane = (n & 15) + (((k >> 3) & 1) << 4);
  int elem = (k & 7) + (((k >> 4) & 1) << 3);
  return ((size_t)(kt * NT + nt) * 32 + lane) * 16 + elem;
}

__device__ __forceinline__ float silu_f(float x)     { return x / (1.f + __expf(-x)); }
__device__ __forceinline__ float softplus_f(float x) { return x > 20.f ? x : log1pf(__expf(x)); }
__device__ __forceinline__ float gelu_f(float x)     { return 0.5f * x * (1.f + erff(x * 0.7071067811865476f)); }

// ---------------------------------------------------------------------------
// K0: convert weights to bf16, fragment-swizzled "B layout" tiles
// ---------------------------------------------------------------------------
__global__ void k_prep_weights(const float* __restrict__ W_in, const float* __restrict__ Wx,
                               const float* __restrict__ Wxb, const float* __restrict__ W_out,
                               const float* __restrict__ W1m, const float* __restrict__ W2m,
                               bf16_t* winb, bf16_t* wxfb, bf16_t* wxbb,
                               bf16_t* woutb, bf16_t* w1b, bf16_t* w2b) {
  int i = blockIdx.x * blockDim.x + threadIdx.x;
  if (i < 128 * 512) {                       // W_in (512x128) -> B[k=128][n=512]
    int k = i >> 9, n = i & 511;
    winb[b_swz(k, n, 32)] = (bf16_t)W_in[n * 128 + k];
  }
  if (i < 256 * XPAD) {                      // Wx (40x256) -> B[k=256][n=48] padded
    int k = i / XPAD, n = i % XPAD;
    wxfb[b_swz(k, n, 3)] = (bf16_t)(n < 40 ? Wx[n * 256 + k] : 0.f);
    wxbb[b_swz(k, n, 3)] = (bf16_t)(n < 40 ? Wxb[n * 256 + k] : 0.f);
  }
  if (i < 256 * 128) {                       // W_out (128x256) -> B[k=256][n=128]
    int k = i >> 7, n = i & 127;
    woutb[b_swz(k, n, 8)] = (bf16_t)W_out[n * 256 + k];
  }
  if (i < 128 * 256) {                       // W1m already [k=128][n=256]
    int k = i >> 8, n = i & 255;
    w1b[b_swz(k, n, 16)] = (bf16_t)W1m[i];
  }
  if (i < 256 * 128) {                       // W2m already [k=256][n=128]
    int k = i >> 7, n = i & 127;
    w2b[b_swz(k, n, 8)] = (bf16_t)W2m[i];
  }
}

// ---------------------------------------------------------------------------
// K1: permute x -> xs, LN1, xz = ln(xs) @ W_in^T   (K=128, N=512)
// ---------------------------------------------------------------------------
__global__ void __launch_bounds__(256)
k_ln1_gemm_win(const float* __restrict__ x,
               const float* __restrict__ g1, const float* __restrict__ b1,
               const bf16_t* __restrict__ winb,
               float* __restrict__ ws_xs,
               bf16_t* __restrict__ ws_xc, bf16_t* __restrict__ ws_z) {
  __shared__ bf16_t aA[4][32][16] __attribute__((aligned(32)));
  __shared__ bf16_t oz[16][512] __attribute__((aligned(32)));
  __shared__ float rsum[16][16], rsq[16][16], rmean[16], rrstd[16];
  int tid = threadIdx.x;
  int r = tid >> 4, c16 = tid & 15;
  int g0 = blockIdx.x * 16;
  int g = g0 + r;
  int s = g >> 8, t = g & 255;
  int bb = s >> 6, nn = s & 63;
  __builtin_prefetch(winb, 0, 2);

  const float4* src4 = (const float4*)(x + (((size_t)(bb * TLEN + t) * NSP + nn) * CDIM));
  float4 va = src4[c16 * 2], vb = src4[c16 * 2 + 1];
  float4* xs4 = (float4*)(ws_xs + (size_t)g * CDIM);
  xs4[c16 * 2] = va; xs4[c16 * 2 + 1] = vb;
  float vals[8] = {va.x, va.y, va.z, va.w, vb.x, vb.y, vb.z, vb.w};
  float ls = 0.f, lq = 0.f;
#pragma unroll
  for (int j = 0; j < 8; ++j) { ls += vals[j]; lq += vals[j] * vals[j]; }
  rsum[r][c16] = ls; rsq[r][c16] = lq;
  __syncthreads();
  if (c16 == 0) {
    float s1 = 0.f, s2 = 0.f;
#pragma unroll
    for (int j = 0; j < 16; ++j) { s1 += rsum[r][j]; s2 += rsq[r][j]; }
    float m = s1 / CDIM, var = s2 / CDIM - m * m;
    rmean[r] = m; rrstd[r] = rsqrtf(var + 1e-5f);
  }
  __syncthreads();
  {
    float m = rmean[r], rs = rrstd[r];
#pragma unroll
    for (int j = 0; j < 8; ++j) {
      int c = c16 * 8 + j;
      int kin = c & 31;
      aA[c >> 5][a_lane(r, kin)][a_elem(kin)] = (bf16_t)((vals[j] - m) * rs * g1[c] + b1[c]);
    }
  }
  __syncthreads();

  int wave = tid >> 5, lane = tid & 31;
#pragma unroll 1
  for (int cb = 0; cb < 4; ++cb) {
    int n0 = cb * 128 + wave * 16;
    int nt = n0 >> 4;
    v8f acc = {};
#pragma unroll
    for (int ks = 0; ks < 4; ++ks) {
      v16bf af = *(const v16bf*)&aA[ks][lane][0];
      v16bf bfr = *(const v16bf*)(winb + (size_t)(ks * 32 + nt) * 512 + lane * 16);
      acc = __builtin_amdgcn_wmma_f32_16x16x32_bf16(false, af, false, bfr, (short)0, acc, false, false);
    }
    int col = n0 + (lane & 15);
    int rowb = (lane >> 4) * 8;
#pragma unroll
    for (int v = 0; v < 8; ++v) oz[rowb + v][col] = (bf16_t)acc[v];
  }
  __syncthreads();
  // coalesced vector writes: 16x512 bf16 -> xc (first 256 cols) / z
#pragma unroll
  for (int i = 0; i < 4; ++i) {
    int e = (tid + i * 256) * 8;
    int rr = e >> 9, c0 = e & 511;
    v8bf w = *(const v8bf*)&oz[rr][c0];
    if (c0 < 256) *(v8bf*)(ws_xc + (size_t)(g0 + rr) * DIN + c0) = w;
    else          *(v8bf*)(ws_z  + (size_t)(g0 + rr) * DIN + (c0 - 256)) = w;
  }
}

// ---------------------------------------------------------------------------
// K2: depthwise causal conv (fwd) + anti-causal conv (bwd) + SiLU, 8 ch/thread
// ---------------------------------------------------------------------------
__global__ void __launch_bounds__(256)
k_conv_silu(const bf16_t* __restrict__ xc,
            const float* __restrict__ cw,  const float* __restrict__ cb,
            const float* __restrict__ cwb, const float* __restrict__ cbb,
            bf16_t* __restrict__ uf, bf16_t* __restrict__ ub) {
  int idx = blockIdx.x * blockDim.x + threadIdx.x;   // TOK * 32
  int d8 = idx & 31;                                  // block of 8 channels
  int g = idx >> 5;
  int t = g & 255, s = g >> 8;
  int c0 = d8 * 8;
  float wf[8][4], wbk[8][4], accf[8], accb[8];
#pragma unroll
  for (int e = 0; e < 8; ++e) {
    float4 t4 = ((const float4*)cw)[c0 + e];
    wf[e][0] = t4.x; wf[e][1] = t4.y; wf[e][2] = t4.z; wf[e][3] = t4.w;
    float4 t5 = ((const float4*)cwb)[c0 + e];
    wbk[e][0] = t5.x; wbk[e][1] = t5.y; wbk[e][2] = t5.z; wbk[e][3] = t5.w;
    accf[e] = cb[c0 + e]; accb[e] = cbb[c0 + e];
  }
  size_t sbase = (size_t)(s << 8) * DIN + c0;
#pragma unroll
  for (int j = 0; j < 4; ++j) {
    int tf = t - 3 + j;
    if (tf >= 0) {
      v8bf xv = *(const v8bf*)(xc + sbase + (size_t)tf * DIN);
#pragma unroll
      for (int e = 0; e < 8; ++e) accf[e] += wf[e][j] * (float)xv[e];
    }
    int tb = t + 3 - j;
    if (tb < TLEN) {
      v8bf xv = *(const v8bf*)(xc + sbase + (size_t)tb * DIN);
#pragma unroll
      for (int e = 0; e < 8; ++e) accb[e] += wbk[e][j] * (float)xv[e];
    }
  }
  v8bf uo, ubo;
#pragma unroll
  for (int e = 0; e < 8; ++e) {
    uo[e]  = (bf16_t)silu_f(accf[e]);
    ubo[e] = (bf16_t)silu_f(accb[e]);
  }
  size_t obase = (size_t)g * DIN + c0;
  *(v8bf*)(uf + obase) = uo;
  *(v8bf*)(ub + obase) = ubo;
}

// ---------------------------------------------------------------------------
// K3: xdbl = u @ Wx^T  (K=256, N=48 padded); 3 waves, A staged swizzled in LDS
// ---------------------------------------------------------------------------
__global__ void __launch_bounds__(96)
k_gemm_wx(const bf16_t* __restrict__ uf, const bf16_t* __restrict__ ub,
          const bf16_t* __restrict__ wxfb, const bf16_t* __restrict__ wxbb,
          float* __restrict__ xdf, float* __restrict__ xdb) {
  __shared__ bf16_t aU[8][32][16] __attribute__((aligned(32)));
  __shared__ float oxd[16][XPAD] __attribute__((aligned(16)));
  const bf16_t* u  = blockIdx.y ? ub   : uf;
  const bf16_t* wb = blockIdx.y ? wxbb : wxfb;
  float*        xo = blockIdx.y ? xdb  : xdf;
  int tid = threadIdx.x;
  int g0 = blockIdx.x * 16;
  __builtin_prefetch(wb, 0, 2);
  const bf16_t* abase = u + (size_t)g0 * DIN;
  for (int v8i = tid; v8i < 512; v8i += 96) {
    int e = v8i * 8;
    int rr = e >> 8, cc0 = e & 255;
    v8bf xv = *(const v8bf*)(abase + (size_t)rr * DIN + cc0);
#pragma unroll
    for (int j = 0; j < 8; ++j) {
      int c = cc0 + j;
      int kin = c & 31;
      aU[c >> 5][a_lane(rr, kin)][a_elem(kin)] = xv[j];
    }
  }
  __syncthreads();
  int wave = tid >> 5, lane = tid & 31;
  int n0 = wave * 16;
  v8f acc = {};
#pragma unroll 1
  for (int ks = 0; ks < 8; ++ks) {
    v16bf af = *(const v16bf*)&aU[ks][lane][0];
    v16bf bfr = *(const v16bf*)(wb + (size_t)(ks * 3 + wave) * 512 + lane * 16);
    acc = __builtin_amdgcn_wmma_f32_16x16x32_bf16(false, af, false, bfr, (short)0, acc, false, false);
  }
  int col = n0 + (lane & 15);
  int rowb = (lane >> 4) * 8;
#pragma unroll
  for (int v = 0; v < 8; ++v) oxd[rowb + v][col] = acc[v];
  __syncthreads();
#pragma unroll
  for (int i = 0; i < 2; ++i) {                 // 16 rows x 12 float4 = 192 / 96
    int q = tid + i * 96;
    int rr = q / 12, c4 = q % 12;
    *(float4*)(xo + (size_t)(g0 + rr) * XPAD + c4 * 4) = *(const float4*)&oxd[rr][c4 * 4];
  }
}

// ---------------------------------------------------------------------------
// K4: selective scan; grid=(256 seqs, 2 dirs); one channel/thread; state in
//     VGPRs; 48-float xdbl row double-buffered through LDS (1 barrier/step)
// ---------------------------------------------------------------------------
__global__ void __launch_bounds__(DIN)
k_scan(const bf16_t* __restrict__ uf, const bf16_t* __restrict__ ub,
       const float* __restrict__ xdf, const float* __restrict__ xdb,
       const float* __restrict__ A_log,   const float* __restrict__ Dv,
       const float* __restrict__ Wdt,     const float* __restrict__ bdt,
       const float* __restrict__ A_b_log, const float* __restrict__ Dv_b,
       const float* __restrict__ Wdtb,    const float* __restrict__ bdtb,
       bf16_t* __restrict__ yf, bf16_t* __restrict__ yb) {
  __shared__ float rowbuf[2][XPAD];
  int d  = threadIdx.x;
  int s  = blockIdx.x;
  int br = blockIdx.y;
  const bf16_t* u  = br ? ub      : uf;
  const float*  xd = br ? xdb     : xdf;
  const float*  Al = br ? A_b_log : A_log;
  const float*  Dp = br ? Dv_b    : Dv;
  const float*  Wd = br ? Wdtb    : Wdt;
  const float*  bd = br ? bdtb    : bdt;
  bf16_t*       yo = br ? yb      : yf;

  float Arow[DST];
#pragma unroll
  for (int n = 0; n < DST; ++n) Arow[n] = -__expf(Al[d * DST + n]);
  float wdt[DTR];
#pragma unroll
  for (int r = 0; r < DTR; ++r) wdt[r] = Wd[d * DTR + r];
  float bdt_d = bd[d];
  float Dd = Dp[d];
  float h[DST];
#pragma unroll
  for (int n = 0; n < DST; ++n) h[n] = 0.f;

  int t0 = br ? (TLEN - 1) : 0;
  if (d < XPAD) rowbuf[0][d] = xd[(size_t)((s << 8) + t0) * XPAD + d];
  __syncthreads();

  for (int step = 0; step < TLEN; ++step) {
    int cur = step & 1;
    int t = br ? (TLEN - 1 - step) : step;
    int g = (s << 8) + t;
    if (step + 1 < TLEN && d < XPAD) {
      int tn = br ? (TLEN - 2 - step) : (step + 1);
      rowbuf[cur ^ 1][d] = xd[(size_t)((s << 8) + tn) * XPAD + d];
    }
    const float* row = rowbuf[cur];
    float dtr = bdt_d;
#pragma unroll
    for (int r = 0; r < DTR; ++r) dtr += row[r] * wdt[r];
    float dt  = softplus_f(dtr);
    float uv  = (float)u[(size_t)g * DIN + d];
    float dtu = dt * uv;
    float y = 0.f;
#pragma unroll
    for (int n = 0; n < DST; ++n) {
      float dA = __expf(dt * Arow[n]);
      h[n] = dA * h[n] + dtu * row[8 + n];
      y += h[n] * row[24 + n];
    }
    yo[(size_t)g * DIN + d] = (bf16_t)(y + uv * Dd);
    __syncthreads();
  }
}

// ---------------------------------------------------------------------------
// K5: A = (y_f + y_b)*silu(z); y = A @ W_out^T (K=256,N=128); LN2; +xs;
//     write xo into d_out in (B,T,N,C) layout.
// ---------------------------------------------------------------------------
__global__ void __launch_bounds__(256)
k_gate_wout_ln2(const bf16_t* __restrict__ yfv, const bf16_t* __restrict__ ybv,
                const bf16_t* __restrict__ zz,  const bf16_t* __restrict__ woutb,
                const float* __restrict__ ws_xs,
                const float* __restrict__ g2, const float* __restrict__ b2,
                float* __restrict__ out) {
  __shared__ bf16_t aY[8][32][16] __attribute__((aligned(32)));
  __shared__ float yo[16][CDIM];
  __shared__ float rsum[16][16], rsq[16][16], rmean[16], rrstd[16];
  int tid = threadIdx.x;
  int g0 = blockIdx.x * 16;
  __builtin_prefetch(woutb, 0, 2);
#pragma unroll
  for (int i = 0; i < 2; ++i) {
    int e = (tid + i * 256) * 8;
    int rr = e >> 8, c0 = e & 255;
    size_t off = (size_t)(g0 + rr) * DIN + c0;
    v8bf y8 = *(const v8bf*)(yfv + off);
    v8bf b8 = *(const v8bf*)(ybv + off);
    v8bf z8 = *(const v8bf*)(zz + off);
#pragma unroll
    for (int j = 0; j < 8; ++j) {
      int c = c0 + j;
      int kin = c & 31;
      float v = ((float)y8[j] + (float)b8[j]) * silu_f((float)z8[j]);
      aY[c >> 5][a_lane(rr, kin)][a_elem(kin)] = (bf16_t)v;
    }
  }
  __syncthreads();

  int wave = tid >> 5, lane = tid & 31;
  int n0 = wave * 16;
  v8f acc = {};
#pragma unroll 1
  for (int ks = 0; ks < 8; ++ks) {
    v16bf af = *(const v16bf*)&aY[ks][lane][0];
    v16bf bfr = *(const v16bf*)(woutb + (size_t)(ks * 8 + wave) * 512 + lane * 16);
    acc = __builtin_amdgcn_wmma_f32_16x16x32_bf16(false, af, false, bfr, (short)0, acc, false, false);
  }
  int col = n0 + (lane & 15);
  int rowb = (lane >> 4) * 8;
#pragma unroll
  for (int v = 0; v < 8; ++v) yo[rowb + v][col] = acc[v];
  __syncthreads();

  int r = tid >> 4, c16 = tid & 15;
  float ls = 0.f, lq = 0.f;
#pragma unroll
  for (int j = 0; j < 8; ++j) { float v = yo[r][c16 * 8 + j]; ls += v; lq += v * v; }
  rsum[r][c16] = ls; rsq[r][c16] = lq;
  __syncthreads();
  if (c16 == 0) {
    float s1 = 0.f, s2 = 0.f;
#pragma unroll
    for (int j = 0; j < 16; ++j) { s1 += rsum[r][j]; s2 += rsq[r][j]; }
    float m = s1 / CDIM, var = s2 / CDIM - m * m;
    rmean[r] = m; rrstd[r] = rsqrtf(var + 1e-5f);
  }
  __syncthreads();
  int g = g0 + r;
  int s = g >> 8, t = g & 255;
  int bb = s >> 6, nn = s & 63;
  float m = rmean[r], rs = rrstd[r];
  const float4* xs4 = (const float4*)(ws_xs + (size_t)g * CDIM);
  float4 xa = xs4[c16 * 2], xb = xs4[c16 * 2 + 1];
  float xres[8] = {xa.x, xa.y, xa.z, xa.w, xb.x, xb.y, xb.z, xb.w};
  float o[8];
#pragma unroll
  for (int j = 0; j < 8; ++j) {
    int c = c16 * 8 + j;
    o[j] = (yo[r][c] - m) * rs * g2[c] + b2[c] + xres[j];
  }
  float4* dst4 = (float4*)(out + ((size_t)(bb * TLEN + t) * NSP + nn) * CDIM);
  dst4[c16 * 2]     = make_float4(o[0], o[1], o[2], o[3]);
  dst4[c16 * 2 + 1] = make_float4(o[4], o[5], o[6], o[7]);
}

// ---------------------------------------------------------------------------
// K6: per-token LN3 + MLP (128 -> 256 gelu -> 128) + residual, in-place d_out
// ---------------------------------------------------------------------------
__global__ void __launch_bounds__(256)
k_ln3_mlp(float* __restrict__ out,
          const float* __restrict__ g3, const float* __restrict__ b3,
          const bf16_t* __restrict__ w1b, const float* __restrict__ b1m,
          const bf16_t* __restrict__ w2b, const float* __restrict__ b2m) {
  __shared__ bf16_t aH[4][32][16] __attribute__((aligned(32)));
  __shared__ bf16_t aG[8][32][16] __attribute__((aligned(32)));
  __shared__ float xrow[16][CDIM];
  __shared__ float ores[16][CDIM];
  __shared__ float rsum[16][16], rsq[16][16], rmean[16], rrstd[16];
  int tid = threadIdx.x;
  int r = tid >> 4, c16 = tid & 15;
  int g0 = blockIdx.x * 16;
  __builtin_prefetch(w1b, 0, 2);
  const float4* src4 = (const float4*)(out + (size_t)(g0 + r) * CDIM);
  float4 va = src4[c16 * 2], vb = src4[c16 * 2 + 1];
  float vals[8] = {va.x, va.y, va.z, va.w, vb.x, vb.y, vb.z, vb.w};
  float ls = 0.f, lq = 0.f;
#pragma unroll
  for (int j = 0; j < 8; ++j) {
    xrow[r][c16 * 8 + j] = vals[j];
    ls += vals[j]; lq += vals[j] * vals[j];
  }
  rsum[r][c16] = ls; rsq[r][c16] = lq;
  __syncthreads();
  if (c16 == 0) {
    float s1 = 0.f, s2 = 0.f;
#pragma unroll
    for (int j = 0; j < 16; ++j) { s1 += rsum[r][j]; s2 += rsq[r][j]; }
    float m = s1 / CDIM, var = s2 / CDIM - m * m;
    rmean[r] = m; rrstd[r] = rsqrtf(var + 1e-5f);
  }
  __syncthreads();
  {
    float m = rmean[r], rs = rrstd[r];
#pragma unroll
    for (int j = 0; j < 8; ++j) {
      int c = c16 * 8 + j;
      int kin = c & 31;
      aH[c >> 5][a_lane(r, kin)][a_elem(kin)] = (bf16_t)((vals[j] - m) * rs * g3[c] + b3[c]);
    }
  }
  __syncthreads();

  int wave = tid >> 5, lane = tid & 31;
  // GEMM1: 16x128 @ 128x256, +bias, exact GeLU -> aG (fragment-swizzled)
#pragma unroll 1
  for (int cb = 0; cb < 2; ++cb) {
    int n0 = cb * 128 + wave * 16;
    int nt = n0 >> 4;
    v8f acc = {};
#pragma unroll
    for (int ks = 0; ks < 4; ++ks) {
      v16bf af = *(const v16bf*)&aH[ks][lane][0];
      v16bf bfr = *(const v16bf*)(w1b + (size_t)(ks * 16 + nt) * 512 + lane * 16);
      acc = __builtin_amdgcn_wmma_f32_16x16x32_bf16(false, af, false, bfr, (short)0, acc, false, false);
    }
    int col = n0 + (lane & 15);
    int rowb = (lane >> 4) * 8;
#pragma unroll
    for (int v = 0; v < 8; ++v) {
      int row = rowb + v;
      int kin = col & 31;
      aG[col >> 5][a_lane(row, kin)][a_elem(kin)] = (bf16_t)gelu_f(acc[v] + b1m[col]);
    }
  }
  __syncthreads();
  // GEMM2: 16x256 @ 256x128, +bias, +residual
  {
    int n0 = wave * 16;
    v8f acc = {};
#pragma unroll 1
    for (int ks = 0; ks < 8; ++ks) {
      v16bf af = *(const v16bf*)&aG[ks][lane][0];
      v16bf bfr = *(const v16bf*)(w2b + (size_t)(ks * 8 + wave) * 512 + lane * 16);
      acc = __builtin_amdgcn_wmma_f32_16x16x32_bf16(false, af, false, bfr, (short)0, acc, false, false);
    }
    int col = n0 + (lane & 15);
    int rowb = (lane >> 4) * 8;
#pragma unroll
    for (int v = 0; v < 8; ++v) {
      int row = rowb + v;
      ores[row][col] = acc[v] + b2m[col] + xrow[row][col];
    }
  }
  __syncthreads();
  float4* dst4 = (float4*)(out + (size_t)(g0 + r) * CDIM);
  dst4[c16 * 2]     = *(const float4*)&ores[r][c16 * 8];
  dst4[c16 * 2 + 1] = *(const float4*)&ores[r][c16 * 8 + 4];
}

// ---------------------------------------------------------------------------
// Host launcher
// ---------------------------------------------------------------------------
extern "C" void kernel_launch(void* const* d_in, const int* in_sizes, int n_in,
                              void* d_out, int out_size, void* d_ws, size_t ws_size,
                              hipStream_t stream) {
  (void)in_sizes; (void)n_in; (void)out_size; (void)ws_size;
  const float* x       = (const float*)d_in[0];
  const float* g1      = (const float*)d_in[1];
  const float* b1      = (const float*)d_in[2];
  const float* W_in    = (const float*)d_in[3];
  const float* conv_w  = (const float*)d_in[4];
  const float* conv_b  = (const float*)d_in[5];
  const float* Wx      = (const float*)d_in[6];
  const float* Wdt     = (const float*)d_in[7];
  const float* bdt     = (const float*)d_in[8];
  const float* A_log   = (const float*)d_in[9];
  const float* Dv      = (const float*)d_in[10];
  const float* conv_wb = (const float*)d_in[11];
  const float* conv_bb = (const float*)d_in[12];
  const float* Wxb     = (const float*)d_in[13];
  const float* Wdtb    = (const float*)d_in[14];
  const float* bdtb    = (const float*)d_in[15];
  const float* A_b_log = (const float*)d_in[16];
  const float* D_b     = (const float*)d_in[17];
  const float* W_out   = (const float*)d_in[18];
  const float* g2      = (const float*)d_in[19];
  const float* b2      = (const float*)d_in[20];
  const float* g3      = (const float*)d_in[21];
  const float* b3      = (const float*)d_in[22];
  const float* W1m     = (const float*)d_in[23];
  const float* b1m     = (const float*)d_in[24];
  const float* W2m     = (const float*)d_in[25];
  const float* b2m     = (const float*)d_in[26];
  float* out = (float*)d_out;

  char* ws = (char*)d_ws;
  size_t off = 0;
  auto walloc = [&](size_t bytes) -> void* {
    void* p = ws + off;
    off = (off + bytes + 255) & ~(size_t)255;
    return p;
  };
  bf16_t* winb  = (bf16_t*)walloc((size_t)128 * 512 * 2);
  bf16_t* wxfb  = (bf16_t*)walloc((size_t)256 * XPAD * 2);
  bf16_t* wxbb  = (bf16_t*)walloc((size_t)256 * XPAD * 2);
  bf16_t* woutb = (bf16_t*)walloc((size_t)256 * 128 * 2);
  bf16_t* w1b   = (bf16_t*)walloc((size_t)128 * 256 * 2);
  bf16_t* w2b   = (bf16_t*)walloc((size_t)256 * 128 * 2);
  float*  ws_xs = (float*) walloc((size_t)TOK * CDIM * 4);   // permuted residual
  bf16_t* ws_xc = (bf16_t*)walloc((size_t)TOK * DIN * 2);    // reused as y_f later
  bf16_t* ws_z  = (bf16_t*)walloc((size_t)TOK * DIN * 2);
  bf16_t* ws_uf = (bf16_t*)walloc((size_t)TOK * DIN * 2);
  bf16_t* ws_ub = (bf16_t*)walloc((size_t)TOK * DIN * 2);
  float*  xdf   = (float*) walloc((size_t)TOK * XPAD * 4);
  float*  xdb   = (float*) walloc((size_t)TOK * XPAD * 4);
  bf16_t* ws_yb = (bf16_t*)walloc((size_t)TOK * DIN * 2);
  bf16_t* ws_yf = ws_xc;   // xc dead after conv kernel

  k_prep_weights<<<256, 256, 0, stream>>>(W_in, Wx, Wxb, W_out, W1m, W2m,
                                          winb, wxfb, wxbb, woutb, w1b, w2b);
  k_ln1_gemm_win<<<TOK / 16, 256, 0, stream>>>(x, g1, b1, winb, ws_xs, ws_xc, ws_z);
  k_conv_silu<<<(TOK * 32) / 256, 256, 0, stream>>>(ws_xc, conv_w, conv_b,
                                                    conv_wb, conv_bb, ws_uf, ws_ub);
  k_gemm_wx<<<dim3(TOK / 16, 2), 96, 0, stream>>>(ws_uf, ws_ub, wxfb, wxbb, xdf, xdb);
  k_scan<<<dim3(SEQ, 2), DIN, 0, stream>>>(ws_uf, ws_ub, xdf, xdb,
                                           A_log, Dv, Wdt, bdt,
                                           A_b_log, D_b, Wdtb, bdtb,
                                           ws_yf, ws_yb);
  k_gate_wout_ln2<<<TOK / 16, 256, 0, stream>>>(ws_yf, ws_yb, ws_z, woutb,
                                                ws_xs, g2, b2, out);
  k_ln3_mlp<<<TOK / 16, 256, 0, stream>>>(out, g3, b3, w1b, b1m, w2b, b2m);
}
